// DeepTemplateMatchingModule_41480794144929
// MI455X (gfx1250) — compile-verified, matrix operand order checked by reading, not run on hardware
//
#include <hip/hip_runtime.h>
#include <hip/hip_bf16.h>

// ---------------------------------------------------------------------------
// DeepTemplateMatching on gfx1250: conv trunk + lin1 + GRU-gate GEMMs +
// alignment GEMMs all via V_WMMA_F32_16X16X32_F16 (fp16 in, fp32 acc).
// Tile staging uses GLOBAL_LOAD_ASYNC_TO_LDS_B128 via inline asm (ASYNCcnt).
// ---------------------------------------------------------------------------

typedef __attribute__((ext_vector_type(16))) _Float16 v16h;
typedef __attribute__((ext_vector_type(8)))  _Float16 v8h;
typedef __attribute__((ext_vector_type(8)))  float    v8f;

// ----- CDNA5 async-to-LDS (inline asm; bypasses builtin addrspace typing) --
#define USE_ASYNC_LDS 1

#if USE_ASYNC_LDS
__device__ __forceinline__ void gload_async_b128(const void* g, void* l) {
  // VDST = LDS byte address (low 32 bits of generic LDS pointer),
  // VADDR = 64-bit global address, GV mode (saddr = off).
  asm volatile("global_load_async_to_lds_b128 %0, %1, off"
               :
               : "v"((unsigned)(size_t)l), "v"((unsigned long long)(size_t)g)
               : "memory");
}
__device__ __forceinline__ void async_wait_all() {
  asm volatile("s_wait_asynccnt 0x0" ::: "memory");
}
#endif

// ----- problem constants ---------------------------------------------------
#define NB   16          // batch
#define NS2  32          // 2*B stacked samples
#define TT   512
#define SEQ  496         // S
#define HID  64
#define GLOBROWS 15872   // 2B * S
#define BRLEN    7936    // B * S (one GRU chain)

// conv shapes
#define H1IN 512
#define W1IN 128
#define H1O  508
#define W1O  124
#define H2O  504
#define W2O  120
#define H3O  500
#define W3O  116
#define PW   58          // pooled width

// ----- workspace layout ----------------------------------------------------
__host__ __device__ constexpr size_t AL(size_t x) { return (x + 255) & ~(size_t)255; }

constexpr size_t SZ_XT  = (size_t)NS2 * H1IN * W1IN * 2;            // f16
constexpr size_t SZ_C1  = (size_t)NS2 * 16 * H1O * W1O * 2;
constexpr size_t SZ_C2  = (size_t)NS2 * 32 * H2O * W2O * 2;
constexpr size_t SZ_C3  = (size_t)NS2 * 64 * H3O * W3O * 2;

constexpr size_t O_XT   = 0;
constexpr size_t O_C1   = AL(O_XT + SZ_XT);
constexpr size_t O_C2   = AL(O_C1 + SZ_C1);
constexpr size_t O_C3   = AL(O_C2 + SZ_C2);
constexpr size_t O_P    = O_C2;                                     // alias: C2 dead after conv3
constexpr size_t O_WP1  = AL(O_C3 + SZ_C3);
constexpr size_t O_WP2  = AL(O_WP1 + 16 * 32 * 2);
constexpr size_t O_WP3  = AL(O_WP2 + 32 * 416 * 2);
constexpr size_t O_L1W  = AL(O_WP3 + (size_t)64 * 800 * 2);
constexpr size_t O_WIH  = AL(O_L1W + (size_t)64 * 3712 * 2);
constexpr size_t O_FEAT = AL(O_WIH + 192 * 64 * 2);                 // f16 (15872 x 64)
constexpr size_t O_GI   = AL(O_FEAT + (size_t)GLOBROWS * 64 * 2);   // f32 (15872 x 192)
constexpr size_t O_HS   = AL(O_GI + (size_t)GLOBROWS * 192 * 4);    // f32 (15872 x 64)
constexpr size_t O_HSH  = AL(O_HS + (size_t)GLOBROWS * 64 * 4);     // f16 (15872 x 64)
constexpr size_t O_SC   = AL(O_HSH + (size_t)GLOBROWS * 64 * 2);    // f32 (16 x 496 x 512)
constexpr size_t O_AS   = AL(O_SC + (size_t)NB * SEQ * 512 * 4);    // f16 (16 x 496 x 512)
constexpr size_t O_TMT  = AL(O_AS + (size_t)NB * SEQ * 512 * 2);    // f16 (16 x 64 x 512)
constexpr size_t O_TP   = AL(O_TMT + (size_t)NB * 64 * 512 * 2);    // f32 (16 x 496 x 64)

// ----- WMMA fragment loader (ISA 7.12.2, 16-bit A 16x32 layout) ------------
__device__ __forceinline__ v16h load_frag16x32(const _Float16* p, int ld) {
  int lane = threadIdx.x & 31;
  int row  = lane & 15;
  int hi   = (lane >> 4) & 1;
  const _Float16* r = p + row * ld;
  v16h f;
#pragma unroll
  for (int e = 0; e < 16; ++e) {
    int v = e >> 1, pb = e & 1;
    int k = (v < 4) ? (hi * 8 + v * 2 + pb) : (16 + hi * 8 + (v - 4) * 2 + pb);
    f[e] = r[k];
  }
  return f;
}

// ===========================================================================
// Packing kernels
// ===========================================================================
__global__ void k_pack_xt(const float* __restrict__ ev, const float* __restrict__ tm,
                          _Float16* __restrict__ xt) {
  size_t i = (size_t)blockIdx.x * 256 + threadIdx.x;
  const size_t total = (size_t)NS2 * H1IN * W1IN;
  if (i >= total) return;
  int c = (int)(i % W1IN);
  size_t r = i / W1IN;
  int t = (int)(r % H1IN);
  int s2 = (int)(r / H1IN);
  const float* src = (s2 < NB) ? ev : tm;
  int b = s2 & 15;
  xt[i] = (_Float16)src[((size_t)b * 128 + c) * TT + t];
}

__global__ void k_pack_w(const float* __restrict__ src, _Float16* __restrict__ dst,
                         int rows, int K, int Kpad) {
  size_t i = (size_t)blockIdx.x * 256 + threadIdx.x;
  size_t total = (size_t)rows * Kpad;
  if (i >= total) return;
  int k = (int)(i % Kpad);
  int r = (int)(i / Kpad);
  dst[i] = (k < K) ? (_Float16)src[(size_t)r * K + k] : (_Float16)0.f;
}

// ===========================================================================
// Implicit-GEMM 5x5 VALID conv, WMMA tensorized.
// GEMM view:  D[oc][pix] = sum_k W[oc][k] * patch[pix][k],  k = (ic,kh,kw)
// ===========================================================================
template <int IC, int OC, int HIN, int WIN>
__global__ __launch_bounds__(256) void k_conv5x5_wmma(const _Float16* __restrict__ in,
                                                      const _Float16* __restrict__ wp,
                                                      const float* __restrict__ bias,
                                                      _Float16* __restrict__ out) {
  constexpr int HOUT = HIN - 4, WOUT = WIN - 4;
  constexpr int MPIX = HOUT * WOUT;
  constexpr int K    = IC * 25;
  constexpr int K32  = (K + 31) / 32;
  constexpr int KPAD = K32 * 32;
  constexpr int NT   = OC / 16;

  __shared__ _Float16 Ps[128][40];  // patches (pixels x k)
  __shared__ _Float16 Ws[64][40];   // weights (oc x k)

  const int smp = blockIdx.y;
  const _Float16* inS = in + (size_t)smp * IC * HIN * WIN;
  const int p0   = blockIdx.x * 128;
  const int tid  = threadIdx.x;
  const int wave = tid >> 5;
  const int lane = tid & 31;

  v8f acc[NT] = {};

  for (int kt = 0; kt < K32; ++kt) {
    const int k0 = kt * 32;
    // stage weights (pre-packed, K zero-padded, contiguous rows)
#if USE_ASYNC_LDS
    if (tid < OC * 4) {
      int oc = tid >> 2, ch = tid & 3;
      gload_async_b128(wp + (size_t)oc * KPAD + k0 + ch * 8, &Ws[oc][ch * 8]);
    }
#else
    for (int i = tid; i < OC * 32; i += 256) {
      int oc = i >> 5, kk = i & 31;
      Ws[oc][kk] = wp[(size_t)oc * KPAD + k0 + kk];
    }
#endif
    // stage patches via im2col gather (2-byte scatter: VALU path)
    for (int i = tid; i < 128 * 32; i += 256) {
      int pl = i >> 5, kk = i & 31;
      int k = k0 + kk;
      int pix = p0 + pl;
      _Float16 v = (_Float16)0.f;
      if (k < K && pix < MPIX) {
        int ic = k / 25, rc = k % 25, kr = rc / 5, kc = rc % 5;
        int h = pix / WOUT, w = pix % WOUT;
        v = inS[((size_t)ic * HIN + h + kr) * WIN + (w + kc)];
      }
      Ps[pl][kk] = v;
    }
#if USE_ASYNC_LDS
    async_wait_all();
#endif
    __syncthreads();

    v16h bf = load_frag16x32(&Ps[wave * 16][0], 40);  // B := patches (N = pixels)
#pragma unroll
    for (int j = 0; j < NT; ++j) {
      v16h af = load_frag16x32(&Ws[j * 16][0], 40);   // A := weights (M = oc)
      acc[j] = __builtin_amdgcn_wmma_f32_16x16x32_f16(false, af, false, bf,
                                                      (short)0, acc[j], false, false);
    }
    __syncthreads();
  }

  const int col = lane & 15, hi = lane >> 4;
  const int pix = p0 + wave * 16 + col;
  _Float16* outS = out + (size_t)smp * OC * MPIX;
  if (pix < MPIX) {
#pragma unroll
    for (int j = 0; j < NT; ++j)
#pragma unroll
      for (int r = 0; r < 8; ++r) {
        int oc = j * 16 + r + hi * 8;
        outS[(size_t)oc * MPIX + pix] = (_Float16)(acc[j][r] + bias[oc]);
      }
  }
}

// ===========================================================================
// Max pool (5,2) stride (1,2) VALID:  (32,64,500,116) -> (32,64,496,58)
// ===========================================================================
__global__ void k_pool(const _Float16* __restrict__ c3, _Float16* __restrict__ p) {
  size_t i = (size_t)blockIdx.x * 256 + threadIdx.x;
  const size_t total = (size_t)NS2 * 64 * SEQ * PW;
  if (i >= total) return;
  int w = (int)(i % PW);
  size_t r0 = i / PW;
  int s = (int)(r0 % SEQ);
  size_t r1 = r0 / SEQ;
  const _Float16* src = c3 + ((size_t)r1 * H3O + s) * W3O + w * 2;
  float m = -1e30f;
#pragma unroll
  for (int di = 0; di < 5; ++di)
#pragma unroll
    for (int dj = 0; dj < 2; ++dj) m = fmaxf(m, (float)src[di * W3O + dj]);
  p[i] = (_Float16)m;
}

// ===========================================================================
// Generic WMMA GEMM:  C[m][n] = sum_k A[m][k] * Bt[n][k] (+bias[n])
// Block: 128 M x 64 N, 8 waves; batched via grid.z + strides.
// A/B tiles staged by GLOBAL_LOAD_ASYNC_TO_LDS_B128; guard rows (m>=M / n>=N)
// pre-zeroed once (async never writes them, zeros persist across K-steps).
// ===========================================================================
__global__ __launch_bounds__(256) void k_wmma_gemm(
    const _Float16* __restrict__ A, long sA, int lda,
    const _Float16* __restrict__ Bt, long sB, int ldb,
    float* __restrict__ C, long sC, int ldc,
    _Float16* __restrict__ Ch, long sCh, int ldch,
    const float* __restrict__ bias, int M, int N, int K32) {
  __shared__ _Float16 As[128][40];
  __shared__ _Float16 Bs[64][40];

  const int z = blockIdx.z;
  const _Float16* Az = A + (size_t)z * sA;
  const _Float16* Bz = Bt + (size_t)z * sB;
  const int m0 = blockIdx.x * 128;
  const int n0 = blockIdx.y * 64;
  const int tid = threadIdx.x;
  const int wave = tid >> 5, lane = tid & 31;
  const int arow = tid >> 1, ach = tid & 1;      // A tile ownership
  const int brow = tid & 63, bch = tid >> 6;     // B tile ownership

#if USE_ASYNC_LDS
  // pre-zero guard rows once; async loads never touch them
  {
    v8h z8 = {};
    if (m0 + arow >= M) {
      *(v8h*)&As[arow][ach * 16]     = z8;
      *(v8h*)&As[arow][ach * 16 + 8] = z8;
    }
    if (n0 + brow >= N) *(v8h*)&Bs[brow][bch * 8] = z8;
  }
#endif

  v8f acc[4] = {};

  for (int kt = 0; kt < K32; ++kt) {
    const int k0 = kt * 32;
#if USE_ASYNC_LDS
    {
      int gm = m0 + arow;
      if (gm < M) {
        const _Float16* g = Az + (size_t)gm * lda + k0 + ach * 16;
        gload_async_b128(g,     &As[arow][ach * 16]);
        gload_async_b128(g + 8, &As[arow][ach * 16 + 8]);
        if (kt + 1 < K32) __builtin_prefetch(g + 32, 0, 1);
      }
      int gn = n0 + brow;
      if (gn < N)
        gload_async_b128(Bz + (size_t)gn * ldb + k0 + bch * 8, &Bs[brow][bch * 8]);
    }
    async_wait_all();
#else
    {
      int c0 = ach * 16;
      int gm = m0 + arow;
      v8h z8 = {};
      v8h v0 = z8, v1 = z8;
      if (gm < M) {
        const _Float16* src = Az + (size_t)gm * lda + k0 + c0;
        v0 = *(const v8h*)(src);
        v1 = *(const v8h*)(src + 8);
        if (kt + 1 < K32) __builtin_prefetch(src + 32, 0, 1);
      }
      *(v8h*)&As[arow][c0]     = v0;
      *(v8h*)&As[arow][c0 + 8] = v1;
    }
    {
      int gn = n0 + brow;
      v8h v = {};
      if (gn < N) v = *(const v8h*)(Bz + (size_t)gn * ldb + k0 + bch * 8);
      *(v8h*)&Bs[brow][bch * 8] = v;
    }
#endif
    __syncthreads();

    v16h af = load_frag16x32(&As[wave * 16][0], 40);
#pragma unroll
    for (int j = 0; j < 4; ++j) {
      v16h bf = load_frag16x32(&Bs[j * 16][0], 40);
      acc[j] = __builtin_amdgcn_wmma_f32_16x16x32_f16(false, af, false, bf,
                                                      (short)0, acc[j], false, false);
    }
    __syncthreads();
  }

  const int col = lane & 15, hi = lane >> 4;
#pragma unroll
  for (int j = 0; j < 4; ++j) {
    int n = n0 + j * 16 + col;
    if (n >= N) continue;
    float bv = bias ? bias[n] : 0.f;
#pragma unroll
    for (int r = 0; r < 8; ++r) {
      int m = m0 + wave * 16 + r + hi * 8;
      if (m >= M) continue;
      float v = acc[j][r] + bv;
      if (C)  C [(size_t)z * sC  + (size_t)m * ldc  + n] = v;
      if (Ch) Ch[(size_t)z * sCh + (size_t)m * ldch + n] = (_Float16)v;
    }
  }
}

// ===========================================================================
// GRU: two sequential chains of 7936 steps (hidden carried across batch,
// as in the reference). 192 threads; W_hh row cached in registers.
// ===========================================================================
__global__ __launch_bounds__(192) void k_gru(const float* __restrict__ gi,
                                             const float* __restrict__ W_hh,
                                             const float* __restrict__ b_hh,
                                             float* __restrict__ hs,
                                             _Float16* __restrict__ hsh) {
  __shared__ float h[HID];
  __shared__ float gh[192];
  const int tid = threadIdx.x;
  float wrow[HID];
#pragma unroll
  for (int j = 0; j < HID; ++j) wrow[j] = W_hh[tid * HID + j];
  const float bh = b_hh[tid];
  if (tid < HID) h[tid] = 0.f;
  __syncthreads();

  const long base = (long)blockIdx.x * BRLEN;
  for (int t = 0; t < BRLEN; ++t) {
    float acc = bh;
#pragma unroll 8
    for (int j = 0; j < HID; ++j) acc += wrow[j] * h[j];
    gh[tid] = acc;
    __syncthreads();
    if (tid < HID) {
      const float* g = gi + (base + t) * 192;
      float r = 1.f / (1.f + __expf(-(g[tid] + gh[tid])));
      float z = 1.f / (1.f + __expf(-(g[HID + tid] + gh[HID + tid])));
      float n = tanhf(g[2 * HID + tid] + r * gh[2 * HID + tid]);
      float hn = (1.f - z) * n + z * h[tid];
      h[tid] = hn;
      hs[(base + t) * HID + tid] = hn;
      hsh[(base + t) * HID + tid] = (_Float16)hn;
    }
    __syncthreads();
  }
}

// ===========================================================================
// Column softmax over m (axis=1) of scores, emit fp16 A_soft (K-padded to 512)
// ===========================================================================
__global__ void k_softmax_col(const float* __restrict__ sc, _Float16* __restrict__ as) {
  const int b = blockIdx.x, n = blockIdx.y, tid = threadIdx.x;
  _Float16* out = as + (size_t)b * SEQ * 512 + n;
  if (n >= SEQ) {
    for (int m = tid; m < SEQ; m += 128) out[(size_t)m * 512] = (_Float16)0.f;
    return;
  }
  const float* col = sc + (size_t)b * SEQ * 512 + n;
  __shared__ float red[128];
  float mx = -1e30f;
  for (int m = tid; m < SEQ; m += 128) mx = fmaxf(mx, col[(size_t)m * 512]);
  red[tid] = mx; __syncthreads();
  for (int s = 64; s > 0; s >>= 1) { if (tid < s) red[tid] = fmaxf(red[tid], red[tid + s]); __syncthreads(); }
  mx = red[0]; __syncthreads();
  float sum = 0.f;
  for (int m = tid; m < SEQ; m += 128) sum += __expf(col[(size_t)m * 512] - mx);
  red[tid] = sum; __syncthreads();
  for (int s = 64; s > 0; s >>= 1) { if (tid < s) red[tid] += red[tid + s]; __syncthreads(); }
  float inv = 1.f / red[0];
  for (int m = tid; m < SEQ; m += 128)
    out[(size_t)m * 512] = (_Float16)(__expf(col[(size_t)m * 512] - mx) * inv);
}

// TmT[b][d][n] = Tm[b][n][d], n padded to 512 with zeros
__global__ void k_build_tmt(const float* __restrict__ hs, _Float16* __restrict__ tmt) {
  size_t i = (size_t)blockIdx.x * 256 + threadIdx.x;
  const size_t total = (size_t)NB * 64 * 512;
  if (i >= total) return;
  int n = (int)(i % 512);
  size_t r = i / 512;
  int d = (int)(r % 64);
  int b = (int)(r / 64);
  tmt[i] = (n < SEQ)
         ? (_Float16)hs[((size_t)BRLEN + (size_t)b * SEQ + n) * HID + d]
         : (_Float16)0.f;
}

// ===========================================================================
// Attention + |Tp-E| pooling + MLP head. One block per sample.
// ===========================================================================
__global__ __launch_bounds__(256) void k_attend_head(
    const float* __restrict__ hs, const float* __restrict__ tp,
    const float* __restrict__ att_w, const float* __restrict__ att_b,
    const float* __restrict__ l3w, const float* __restrict__ l3b,
    const float* __restrict__ cw, const float* __restrict__ cb,
    float* __restrict__ out) {
  const int b = blockIdx.x, tid = threadIdx.x;
  __shared__ float l[SEQ];
  __shared__ float red[256];
  __shared__ float aw[HID];
  __shared__ float rep[HID];
  __shared__ float h2[128];
  if (tid < HID) aw[tid] = att_w[tid];
  __syncthreads();
  const float* E  = hs + (size_t)b * SEQ * HID;
  const float* Tp = tp + (size_t)b * SEQ * HID;
  for (int s = tid; s < SEQ; s += 256) {
    float acc = 0.f;
    for (int d = 0; d < HID; ++d) acc += E[(size_t)s * HID + d] * aw[d];
    l[s] = acc + att_b[0];
  }
  __syncthreads();
  float mx = -1e30f;
  for (int s = tid; s < SEQ; s += 256) mx = fmaxf(mx, l[s]);
  red[tid] = mx; __syncthreads();
  for (int st = 128; st > 0; st >>= 1) { if (tid < st) red[tid] = fmaxf(red[tid], red[tid + st]); __syncthreads(); }
  mx = red[0]; __syncthreads();
  float sm = 0.f;
  for (int s = tid; s < SEQ; s += 256) { float e = __expf(l[s] - mx); l[s] = e; sm += e; }
  red[tid] = sm; __syncthreads();
  for (int st = 128; st > 0; st >>= 1) { if (tid < st) red[tid] += red[tid + st]; __syncthreads(); }
  const float inv = 1.f / red[0];
  __syncthreads();
  if (tid < HID) {
    float acc = 0.f;
    for (int s = 0; s < SEQ; ++s)
      acc += l[s] * inv * fabsf(Tp[(size_t)s * HID + tid] - E[(size_t)s * HID + tid]);
    rep[tid] = fmaxf(acc, 0.f);
  }
  __syncthreads();
  if (tid < 128) {
    float acc = l3b[tid];
    for (int d = 0; d < HID; ++d) acc += rep[d] * l3w[tid * HID + d];
    h2[tid] = fmaxf(acc, 0.f);
  }
  __syncthreads();
  if (tid < 2) {
    float acc = cb[tid];
    for (int j = 0; j < 128; ++j) acc += h2[j] * cw[tid * 128 + j];
    out[b * 2 + tid] = acc;
  }
}

// ===========================================================================
// Host launcher
// ===========================================================================
static inline int cdiv_i(size_t a, size_t b) { return (int)((a + b - 1) / b); }

extern "C" void kernel_launch(void* const* d_in, const int* in_sizes, int n_in,
                              void* d_out, int out_size, void* d_ws, size_t ws_size,
                              hipStream_t stream) {
  (void)in_sizes; (void)n_in; (void)out_size; (void)ws_size;
  const float* evaluation = (const float*)d_in[0];
  const float* templ      = (const float*)d_in[1];
  const float* conv1_w = (const float*)d_in[2];  const float* conv1_b = (const float*)d_in[3];
  const float* conv2_w = (const float*)d_in[4];  const float* conv2_b = (const float*)d_in[5];
  const float* conv3_w = (const float*)d_in[6];  const float* conv3_b = (const float*)d_in[7];
  const float* lin1_w  = (const float*)d_in[8];  const float* lin1_b  = (const float*)d_in[9];
  const float* W_ih    = (const float*)d_in[10]; const float* W_hh    = (const float*)d_in[11];
  const float* b_ih    = (const float*)d_in[12]; const float* b_hh    = (const float*)d_in[13];
  const float* att_w   = (const float*)d_in[14]; const float* att_b   = (const float*)d_in[15];
  const float* lin3_w  = (const float*)d_in[16]; const float* lin3_b  = (const float*)d_in[17];
  const float* cls_w   = (const float*)d_in[18]; const float* cls_b   = (const float*)d_in[19];
  float* out = (float*)d_out;

  char* ws = (char*)d_ws;
  _Float16* XT   = (_Float16*)(ws + O_XT);
  _Float16* C1   = (_Float16*)(ws + O_C1);
  _Float16* C2   = (_Float16*)(ws + O_C2);
  _Float16* C3   = (_Float16*)(ws + O_C3);
  _Float16* P    = (_Float16*)(ws + O_P);
  _Float16* WP1  = (_Float16*)(ws + O_WP1);
  _Float16* WP2  = (_Float16*)(ws + O_WP2);
  _Float16* WP3  = (_Float16*)(ws + O_WP3);
  _Float16* L1W  = (_Float16*)(ws + O_L1W);
  _Float16* WIH  = (_Float16*)(ws + O_WIH);
  _Float16* FEAT = (_Float16*)(ws + O_FEAT);
  float*    GI   = (float*)   (ws + O_GI);
  float*    HS   = (float*)   (ws + O_HS);
  _Float16* HSH  = (_Float16*)(ws + O_HSH);
  float*    SC   = (float*)   (ws + O_SC);
  _Float16* AS   = (_Float16*)(ws + O_AS);
  _Float16* TMT  = (_Float16*)(ws + O_TMT);
  float*    TP   = (float*)   (ws + O_TP);

  // ---- input transpose/stack + weight packs -------------------------------
  k_pack_xt<<<cdiv_i((size_t)NS2 * H1IN * W1IN, 256), 256, 0, stream>>>(evaluation, templ, XT);
  k_pack_w<<<cdiv_i(16 * 32, 256), 256, 0, stream>>>(conv1_w, WP1, 16, 25, 32);
  k_pack_w<<<cdiv_i(32 * 416, 256), 256, 0, stream>>>(conv2_w, WP2, 32, 400, 416);
  k_pack_w<<<cdiv_i(64 * 800, 256), 256, 0, stream>>>(conv3_w, WP3, 64, 800, 800);
  k_pack_w<<<cdiv_i((size_t)64 * 3712, 256), 256, 0, stream>>>(lin1_w, L1W, 64, 3712, 3712);
  k_pack_w<<<cdiv_i(192 * 64, 256), 256, 0, stream>>>(W_ih, WIH, 192, 64, 64);

  // ---- conv trunk (implicit GEMM, WMMA) -----------------------------------
  k_conv5x5_wmma<1, 16, H1IN, W1IN><<<dim3(cdiv_i((size_t)H1O * W1O, 128), NS2), 256, 0, stream>>>(XT, WP1, conv1_b, C1);
  k_conv5x5_wmma<16, 32, H1O, W1O><<<dim3(cdiv_i((size_t)H2O * W2O, 128), NS2), 256, 0, stream>>>(C1, WP2, conv2_b, C2);
  k_conv5x5_wmma<32, 64, H2O, W2O><<<dim3(cdiv_i((size_t)H3O * W3O, 128), NS2), 256, 0, stream>>>(C2, WP3, conv3_b, C3);

  // ---- pool (writes into dead C2 region) ----------------------------------
  k_pool<<<cdiv_i((size_t)NS2 * 64 * SEQ * PW, 256), 256, 0, stream>>>(C3, P);

  // ---- lin1: (15872 x 3712) x (3712 -> 64), fp16 out ----------------------
  k_wmma_gemm<<<dim3(GLOBROWS / 128, 1, 1), 256, 0, stream>>>(
      P, 0, 3712, L1W, 0, 3712,
      (float*)nullptr, 0, 0, FEAT, 0, 64,
      lin1_b, GLOBROWS, 64, 3712 / 32);

  // ---- gi = feat @ W_ih^T + b_ih: (15872 x 64) x (64 -> 192), fp32 out ----
  k_wmma_gemm<<<dim3(GLOBROWS / 128, 3, 1), 256, 0, stream>>>(
      FEAT, 0, 64, WIH, 0, 64,
      GI, 0, 192, (_Float16*)nullptr, 0, 0,
      b_ih, GLOBROWS, 192, 2);

  // ---- GRU: two serial chains (eval / template) ---------------------------
  k_gru<<<2, 192, 0, stream>>>(GI, W_hh, b_hh, HS, HSH);

  // ---- alignment scores: per-sample (496 x 64) x (496 x 64)^T -------------
  k_wmma_gemm<<<dim3(4, 8, NB), 256, 0, stream>>>(
      HSH, (long)SEQ * HID, HID,
      HSH + (size_t)BRLEN * HID, (long)SEQ * HID, HID,
      SC, (long)SEQ * 512, 512, (_Float16*)nullptr, 0, 0,
      (const float*)nullptr, SEQ, SEQ, 2);

  // ---- softmax over m (column-wise), K-padded fp16 ------------------------
  k_softmax_col<<<dim3(NB, 512), 128, 0, stream>>>(SC, AS);
  k_build_tmt<<<cdiv_i((size_t)NB * 64 * 512, 256), 256, 0, stream>>>(HS, TMT);

  // ---- Tp = A_soft @ Tm: per-sample (496 x 512pad) x (64 x 512pad)^T ------
  k_wmma_gemm<<<dim3(4, 1, NB), 256, 0, stream>>>(
      AS, (long)SEQ * 512, 512,
      TMT, (long)64 * 512, 512,
      TP, (long)SEQ * HID, HID, (_Float16*)nullptr, 0, 0,
      (const float*)nullptr, SEQ, HID, 16);

  // ---- attention pooling + MLP head ---------------------------------------
  k_attend_head<<<NB, 256, 0, stream>>>(HS, TP, att_w, att_b, lin3_w, lin3_b,
                                        cls_w, cls_b, out);
}
